// PatchCore_74990128988401
// MI455X (gfx1250) — matrix operand order repaired and checked by reading, not compiled, not run on hardware
//
#include <hip/hip_runtime.h>
#include <hip/hip_bf16.h>
#include <math.h>

typedef __attribute__((ext_vector_type(16))) _Float16 v16h;
typedef __attribute__((ext_vector_type(8)))  float    v8f;
typedef __attribute__((ext_vector_type(4)))  int      v4i;
typedef unsigned long long ull;
typedef unsigned int       u32;

#define HW28   28
#define M_ROWS 784          // 28*28
#define CCH    1536         // 512 + 1024 channels
#define NLIB   100000
#define KT     48           // 1536 / 32
#define MTILES 49           // 784 / 16
#define NTILES 6250         // 100000 / 16
#define IMG    224

#define STAGE_HALFS 7168    // 7 mtiles * 2 (hi/lo) * 512 halfs = 14336 bytes
#define STAGE_B128  896     // 14336 / 16

// ---------------------------------------------------------------------------
// async global->LDS copy primitive (CDNA5): 16 bytes per lane
// ---------------------------------------------------------------------------
#if defined(__HIP_DEVICE_COMPILE__) && __has_builtin(__builtin_amdgcn_global_load_async_to_lds_b128)
#define HAVE_ASYNC_BUILTIN 1
#else
#define HAVE_ASYNC_BUILTIN 0
#endif

__device__ __forceinline__ void async_cp16(const _Float16* g, _Float16* l) {
#if HAVE_ASYNC_BUILTIN
    __builtin_amdgcn_global_load_async_to_lds_b128(
        (__attribute__((address_space(1))) v4i*)g,
        (__attribute__((address_space(3))) v4i*)l, 0, 0);
#elif defined(__HIP_DEVICE_COMPILE__)
    u32 loff = (u32)(size_t)(__attribute__((address_space(3))) void*)l;
    asm volatile("global_load_async_to_lds_b128 %0, %1, off"
                 :: "v"(loff), "v"((unsigned long long)(size_t)g)
                 : "memory");
#else
    (void)g; (void)l;
#endif
}

__device__ __forceinline__ void async_wait0() {
#if defined(__HIP_DEVICE_COMPILE__) && __has_builtin(__builtin_amdgcn_s_wait_asynccnt)
    __builtin_amdgcn_s_wait_asynccnt(0);
#elif defined(__HIP_DEVICE_COMPILE__)
    asm volatile("s_wait_asynccnt 0x0" ::: "memory");
#endif
}

// ---------------------------------------------------------------------------
// 0. init min-keys
// ---------------------------------------------------------------------------
__global__ void k_init(ull* __restrict__ minkey) {
    int i = blockIdx.x * blockDim.x + threadIdx.x;
    if (i < M_ROWS) minkey[i] = ~0ull;
}

// ---------------------------------------------------------------------------
// 1. patch construction: avg3(VALID) + adaptive pool to 28x28, channel concat
// ---------------------------------------------------------------------------
__global__ void k_patch(const float* __restrict__ f1, const float* __restrict__ f2,
                        float* __restrict__ patch) {
    int m  = blockIdx.x;                 // 0..783
    int oy = m / HW28, ox = m % HW28;
    for (int c = threadIdx.x; c < CCH; c += blockDim.x) {
        const float* f; int nin, W, cc;
        if (c < 512) { f = f1; nin = 26; W = 28; cc = c; }
        else         { f = f2; nin = 12; W = 14; cc = c - 512; }
        int ys = (oy * nin) / HW28, ye = ((oy + 1) * nin + HW28 - 1) / HW28;
        int xs = (ox * nin) / HW28, xe = ((ox + 1) * nin + HW28 - 1) / HW28;
        float acc = 0.f;
        for (int y = ys; y < ye; ++y)
            for (int x = xs; x < xe; ++x) {
                float s = 0.f;
                for (int dy = 0; dy < 3; ++dy)
                    for (int dx = 0; dx < 3; ++dx)
                        s += f[(cc * W + (y + dy)) * W + (x + dx)];
                acc += s * (1.f / 9.f);
            }
        acc /= (float)((ye - ys) * (xe - xs));
        patch[(size_t)m * CCH + c] = acc;
    }
}

// ---------------------------------------------------------------------------
// 2. pack patch into WMMA A-operand layout, f16 hi/lo split, K-tile-major:
//    apk[((kt*49 + mtile)*2 + hl)*512 + lane*16 + j]
//    lane<16:  M=lane,    j<8 -> K=kt*32+j,   j>=8 -> K=kt*32+8+j
//    lane>=16: M=lane-16, j<8 -> K=kt*32+8+j, j>=8 -> K=kt*32+16+j
// ---------------------------------------------------------------------------
__global__ void k_pack(const float* __restrict__ patch, _Float16* __restrict__ apk) {
    int gid = blockIdx.x * 256 + threadIdx.x;
    if (gid >= MTILES * KT * 32 * 16) return;
    int j     = gid & 15;
    int t     = gid >> 4;
    int lane  = t & 31;
    int u     = t >> 5;
    int kt    = u % KT;
    int mtile = u / KT;
    int m     = mtile * 16 + (lane & 15);
    int khalf = lane >> 4;
    int k     = kt * 32 + khalf * 8 + (j < 8 ? j : j + 8);
    float x = patch[(size_t)m * CCH + k];
    _Float16 h = (_Float16)x;
    _Float16 l = (_Float16)(x - (float)h);
    size_t dst = ((size_t)(kt * MTILES + mtile) * 2) * 512 + lane * 16 + j;
    apk[dst]       = h;   // hl = 0
    apk[dst + 512] = l;   // hl = 1
}

// ---------------------------------------------------------------------------
// 3. row squared-norms (wave per row)
// ---------------------------------------------------------------------------
__global__ void k_rownorm(const float* __restrict__ rows, int nrows,
                          float* __restrict__ norms) {
    int wave = threadIdx.x >> 5, lane = threadIdx.x & 31;
    int r = blockIdx.x * 8 + wave;
    if (r >= nrows) return;
    const float* p = rows + (size_t)r * CCH;
    float s = 0.f;
    for (int k = lane; k < CCH; k += 32) { float v = p[k]; s += v * v; }
    for (int off = 16; off; off >>= 1) s += __shfl_xor(s, off);
    if (lane == 0) norms[r] = s;
}

// ---------------------------------------------------------------------------
// 4. WMMA distance kernel.
//    Block = 8 waves; wave w owns N-tile blockIdx.x*8+w; all waves share
//    mgroup = blockIdx.y (7 M-tiles). A operands are staged per K-tile into
//    LDS with async global->LDS copies (double buffered, overlapped with the
//    21 WMMAs of the previous K-tile). The M-tile loop is software-pipelined
//    so ds_loads for mi+1 issue before the WMMAs of mi. d2 = pnorm + lnorm
//    - 2*dot; lane-reduced (d2bits<<32|n) keys -> one atomic-min per row.
// ---------------------------------------------------------------------------
__global__ void __launch_bounds__(256)
k_dist(const _Float16* __restrict__ apk,
       const float* __restrict__ lib, const float* __restrict__ pnorm,
       const float* __restrict__ lnorm, ull* __restrict__ minkey) {
    __shared__ _Float16 stage[2 * STAGE_HALFS];          // 28672 B

    const int tid   = threadIdx.x;
    const int wave  = tid >> 5, lane = tid & 31;
    const int ntile = blockIdx.x * 8 + wave;
    const bool active = (ntile < NTILES);                // wave-uniform
    const int col = lane & 15, khalf = lane >> 4;
    const int nbase  = ntile < NTILES ? ntile * 16 : 0;
    const int mgroup = blockIdx.y;                       // 0..6
    const float* brow = lib + (size_t)(nbase + col) * CCH + khalf * 8;

    // contiguous 14336-B block of packed A for (kt, mgroup):
    const _Float16* agbase = apk + (size_t)mgroup * 7 * 1024;   // + kt*49*1024

    v8f acc[7];
    for (int mi = 0; mi < 7; ++mi)
        for (int r = 0; r < 8; ++r) acc[mi][r] = 0.f;

    // prologue: stage kt=0 into buffer 0
    for (int i = tid; i < STAGE_B128; i += 256)
        async_cp16(agbase + i * 8, stage + i * 8);

    for (int kt = 0; kt < KT; ++kt) {
        async_wait0();          // own async copies for buffer kt&1 done
        __syncthreads();        // everyone's copies done; prev buffer free
        if (kt + 1 < KT) {      // stage next K-tile into the other buffer
            const _Float16* g = agbase + (size_t)(kt + 1) * MTILES * 1024;
            _Float16* l = stage + ((kt + 1) & 1) * STAGE_HALFS;
            for (int i = tid; i < STAGE_B128; i += 256)
                async_cp16(g + i * 8, l + i * 8);
        }
        if (active) {
            const float* bp = brow + kt * 32;
            float4 q0 = *(const float4*)(bp + 0);
            float4 q1 = *(const float4*)(bp + 4);
            float4 q2 = *(const float4*)(bp + 16);
            float4 q3 = *(const float4*)(bp + 20);
            float fv[16] = {q0.x, q0.y, q0.z, q0.w, q1.x, q1.y, q1.z, q1.w,
                            q2.x, q2.y, q2.z, q2.w, q3.x, q3.y, q3.z, q3.w};
            v16h bhi, blo;
            #pragma unroll
            for (int j = 0; j < 16; ++j) {
                _Float16 h = (_Float16)fv[j];
                bhi[j] = h;
                blo[j] = (_Float16)(fv[j] - (float)h);
            }
            if (kt + 1 < KT) __builtin_prefetch(bp + 32, 0, 1);
            const _Float16* sb = stage + (kt & 1) * STAGE_HALFS + lane * 16;
            // software-pipelined M-tile loop (loads for mi+1 before WMMAs of mi)
            v16h aH = *(const v16h*)(sb + 0);
            v16h aL = *(const v16h*)(sb + 512);
            #pragma unroll
            for (int mi = 0; mi < 7; ++mi) {
                v16h nH, nL;
                if (mi < 6) {
                    nH = *(const v16h*)(sb + (mi * 2 + 2) * 512);
                    nL = *(const v16h*)(sb + (mi * 2 + 3) * 512);
                }
                acc[mi] = __builtin_amdgcn_wmma_f32_16x16x32_f16(
                    false, aH, false, bhi, (short)0, acc[mi], false, false);
                acc[mi] = __builtin_amdgcn_wmma_f32_16x16x32_f16(
                    false, aH, false, blo, (short)0, acc[mi], false, false);
                acc[mi] = __builtin_amdgcn_wmma_f32_16x16x32_f16(
                    false, aL, false, bhi, (short)0, acc[mi], false, false);
                if (mi < 6) { aH = nH; aL = nL; }
            }
        }
    }

    if (!active) return;
    const float lnv = lnorm[nbase + col];
    for (int mi = 0; mi < 7; ++mi) {
        int mtile = mgroup * 7 + mi;
        for (int r = 0; r < 8; ++r) {
            int m = mtile * 16 + r + 8 * khalf;   // C layout: vgpr r, half-group khalf
            float d2 = pnorm[m] + lnv - 2.0f * acc[mi][r];
            d2 = fmaxf(d2, 0.0f);
            ull key = ((ull)__float_as_uint(d2) << 32) | (u32)(nbase + col);
            for (int off = 1; off < 16; off <<= 1) {   // min across 16 N-columns
                u32 klo = (u32)key, khi = (u32)(key >> 32);
                u32 olo = __shfl_xor(klo, off);
                u32 ohi = __shfl_xor(khi, off);
                ull ok = ((ull)ohi << 32) | olo;
                if (ok < key) key = ok;
            }
            if (col == 0) atomicMin(minkey + m, key);
        }
    }
}

// ---------------------------------------------------------------------------
// 5. min_val/min_idx unpack + argmax(min_val) -> s_idx, s_star, r
// ---------------------------------------------------------------------------
__global__ void k_argmax(const ull* __restrict__ minkey, float* __restrict__ minval,
                         float* __restrict__ scal_f, int* __restrict__ scal_i) {
    __shared__ ull sk[256];
    int tid = threadIdx.x;
    ull best = 0;
    for (int m = tid; m < M_ROWS; m += 256) {
        ull key = minkey[m];
        u32 bits = (u32)(key >> 32);
        minval[m] = sqrtf(fmaxf(__uint_as_float(bits), 1e-12f));
        ull cand = ((ull)bits << 32) | (u32)(0xFFFFFFFFu - (u32)m); // ties -> low m
        if (cand > best) best = cand;
    }
    sk[tid] = best; __syncthreads();
    for (int s = 128; s; s >>= 1) {
        if (tid < s && sk[tid + s] > sk[tid]) sk[tid] = sk[tid + s];
        __syncthreads();
    }
    if (tid == 0) {
        ull b = sk[0];
        int sidx = (int)(0xFFFFFFFFu - (u32)b);
        scal_f[0] = sqrtf(fmaxf(__uint_as_float((u32)(b >> 32)), 1e-12f)); // s_star
        scal_i[1] = sidx;
        scal_i[2] = (int)(u32)(minkey[sidx] & 0xFFFFFFFFu);                // r
    }
}

// ---------------------------------------------------------------------------
// 6. w_dist^2 = || lib[r] - lib[n] ||^2 (wave per n)
// ---------------------------------------------------------------------------
__global__ void k_wdist(const float* __restrict__ lib, const int* __restrict__ scal_i,
                        float* __restrict__ wd2) {
    int wave = threadIdx.x >> 5, lane = threadIdx.x & 31;
    int n = blockIdx.x * 8 + wave;
    if (n >= NLIB) return;
    const float* a = lib + (size_t)scal_i[2] * CCH;
    const float* b = lib + (size_t)n * CCH;
    float s = 0.f;
    for (int k = lane; k < CCH; k += 32) { float d = a[k] - b[k]; s += d * d; }
    for (int off = 16; off; off >>= 1) s += __shfl_xor(s, off);
    if (lane == 0) wd2[n] = s;
}

// ---------------------------------------------------------------------------
// 7. top-3 smallest of w_dist, knn distances, final scalar s -> out[0]
// ---------------------------------------------------------------------------
__global__ void k_final(const float* __restrict__ wd2, const float* __restrict__ patch,
                        const float* __restrict__ lib, const float* __restrict__ scal_f,
                        const int* __restrict__ scal_i, float* __restrict__ out) {
    __shared__ ull lk[768];
    __shared__ float red[256];
    __shared__ int nnsh[2];
    int tid = threadIdx.x;
    ull b0 = ~0ull, b1 = ~0ull, b2 = ~0ull;
    for (int n = tid; n < NLIB; n += 256) {
        ull key = ((ull)__float_as_uint(wd2[n]) << 32) | (u32)n;
        if (key < b2) {
            b2 = key;
            if (b2 < b1) { ull t = b1; b1 = b2; b2 = t;
                if (b1 < b0) { ull v = b0; b0 = b1; b1 = v; } }
        }
    }
    lk[tid * 3 + 0] = b0; lk[tid * 3 + 1] = b1; lk[tid * 3 + 2] = b2;
    __syncthreads();
    if (tid == 0) {
        ull g0 = ~0ull, g1 = ~0ull, g2 = ~0ull;
        for (int i = 0; i < 768; ++i) {
            ull key = lk[i];
            if (key < g2) {
                g2 = key;
                if (g2 < g1) { ull t = g1; g1 = g2; g2 = t;
                    if (g1 < g0) { ull v = g0; g0 = g1; g1 = v; } }
            }
        }
        nnsh[0] = (int)(u32)(g1 & 0xFFFFFFFFu);   // nn_idx[1]
        nnsh[1] = (int)(u32)(g2 & 0xFFFFFFFFu);   // nn_idx[2]
    }
    __syncthreads();
    const float* mt = patch + (size_t)scal_i[1] * CCH;
    float dk[2];
    for (int t = 0; t < 2; ++t) {
        const float* lb = lib + (size_t)nnsh[t] * CCH;
        float s = 0.f;
        for (int k = tid; k < CCH; k += 256) { float d = mt[k] - lb[k]; s += d * d; }
        red[tid] = s; __syncthreads();
        for (int st = 128; st; st >>= 1) {
            if (tid < st) red[tid] += red[tid + st];
            __syncthreads();
        }
        dk[t] = sqrtf(red[0]);
        __syncthreads();
    }
    if (tid == 0) {
        float sstar = scal_f[0];
        float D = sqrtf((float)CCH);
        float w = 1.0f - expf(sstar / D) / (expf(dk[0] / D) + expf(dk[1] / D));
        out[0] = w * sstar;
    }
}

// ---------------------------------------------------------------------------
// 8. bilinear resize 28x28 -> 224x224 (half-pixel centers, edge clamp)
// ---------------------------------------------------------------------------
__global__ void k_resize(const float* __restrict__ mv, float* __restrict__ outp) {
    int idx = blockIdx.x * 256 + threadIdx.x;
    if (idx >= IMG * IMG) return;
    int oy = idx / IMG, ox = idx % IMG;
    float sy = (oy + 0.5f) * 0.125f - 0.5f;
    float sx = (ox + 0.5f) * 0.125f - 0.5f;
    int y0 = (int)floorf(sy), x0 = (int)floorf(sx);
    float fy = sy - (float)y0, fx = sx - (float)x0;
    int y0c = min(max(y0, 0), 27), y1c = min(max(y0 + 1, 0), 27);
    int x0c = min(max(x0, 0), 27), x1c = min(max(x0 + 1, 0), 27);
    float v00 = mv[y0c * 28 + x0c], v01 = mv[y0c * 28 + x1c];
    float v10 = mv[y1c * 28 + x0c], v11 = mv[y1c * 28 + x1c];
    float v0 = v00 + fx * (v01 - v00);
    float v1 = v10 + fx * (v11 - v10);
    outp[idx] = v0 + fy * (v1 - v0);
}

// ---------------------------------------------------------------------------
// 9. separable 33-tap Gaussian blur (sigma=4), zero 'SAME' padding
// ---------------------------------------------------------------------------
__device__ __forceinline__ void gauss_weights(float* g) {
    if (threadIdx.x == 0) {
        double tmp[33], s = 0.0;
        for (int j = 0; j < 33; ++j) {
            double ax = (double)j - 16.0;
            tmp[j] = exp(-(ax * ax) / 32.0);
            s += tmp[j];
        }
        for (int j = 0; j < 33; ++j) g[j] = (float)(tmp[j] / s);
    }
    __syncthreads();
}

__global__ void k_blur_h(const float* __restrict__ in, float* __restrict__ outp) {
    __shared__ float g[33];
    gauss_weights(g);
    int idx = blockIdx.x * 256 + threadIdx.x;
    if (idx >= IMG * IMG) return;
    int y = idx / IMG, x = idx % IMG;
    float s = 0.f;
    for (int j = 0; j < 33; ++j) {
        int xx = x + j - 16;
        if (xx >= 0 && xx < IMG) s += g[j] * in[y * IMG + xx];
    }
    outp[idx] = s;
}

__global__ void k_blur_v(const float* __restrict__ in, float* __restrict__ outp) {
    __shared__ float g[33];
    gauss_weights(g);
    int idx = blockIdx.x * 256 + threadIdx.x;
    if (idx >= IMG * IMG) return;
    int y = idx / IMG, x = idx % IMG;
    float s = 0.f;
    for (int j = 0; j < 33; ++j) {
        int yy = y + j - 16;
        if (yy >= 0 && yy < IMG) s += g[j] * in[yy * IMG + x];
    }
    outp[idx] = s;
}

// ---------------------------------------------------------------------------
extern "C" void kernel_launch(void* const* d_in, const int* in_sizes, int n_in,
                              void* d_out, int out_size, void* d_ws, size_t ws_size,
                              hipStream_t stream) {
    const float* f1  = (const float*)d_in[0];   // [1,512,28,28]
    const float* f2  = (const float*)d_in[1];   // [1,1024,14,14]
    const float* lib = (const float*)d_in[2];   // [100000,1536]
    float* out = (float*)d_out;                 // [0]=s, [1..50176]=s_map

    char* ws = (char*)d_ws;
    size_t off = 0;
    auto alloc = [&](size_t bytes) -> char* {
        char* p = ws + off;
        off = (off + bytes + 255) & ~(size_t)255;
        return p;
    };
    float*     patch   = (float*)alloc((size_t)M_ROWS * CCH * 4);          // 4.82 MB
    _Float16*  apk     = (_Float16*)alloc((size_t)M_ROWS * CCH * 2 * 2);   // 4.82 MB
    float*     pnorm   = (float*)alloc(M_ROWS * 4);
    float*     lnorm   = (float*)alloc((size_t)NLIB * 4);
    ull*       minkey  = (ull*)alloc(M_ROWS * 8);
    float*     minval  = (float*)alloc(M_ROWS * 4);
    float*     scal_f  = (float*)alloc(64);
    int*       scal_i  = (int*)scal_f;
    float*     wd2     = (float*)alloc((size_t)NLIB * 4);
    float*     resized = (float*)alloc((size_t)IMG * IMG * 4);
    float*     tmpb    = (float*)alloc((size_t)IMG * IMG * 4);
    (void)ws_size; (void)in_sizes; (void)n_in; (void)out_size;

    k_init   <<<4, 256, 0, stream>>>(minkey);
    k_patch  <<<M_ROWS, 256, 0, stream>>>(f1, f2, patch);
    k_pack   <<<(MTILES * KT * 32 * 16) / 256, 256, 0, stream>>>(patch, apk);
    k_rownorm<<<98, 256, 0, stream>>>(patch, M_ROWS, pnorm);
    k_rownorm<<<NLIB / 8, 256, 0, stream>>>(lib, NLIB, lnorm);
    k_dist   <<<dim3((NTILES + 7) / 8, 7), 256, 0, stream>>>(apk, lib, pnorm,
                                                             lnorm, minkey);
    k_argmax <<<1, 256, 0, stream>>>(minkey, minval, scal_f, scal_i);
    k_wdist  <<<NLIB / 8, 256, 0, stream>>>(lib, scal_i, wd2);
    k_final  <<<1, 256, 0, stream>>>(wd2, patch, lib, scal_f, scal_i, out);
    k_resize <<<(IMG * IMG) / 256, 256, 0, stream>>>(minval, resized);
    k_blur_h <<<(IMG * IMG) / 256, 256, 0, stream>>>(resized, tmpb);
    k_blur_v <<<(IMG * IMG) / 256, 256, 0, stream>>>(tmpb, out + 1);
}
